// RationaleExtractor_40810779247516
// MI455X (gfx1250) — matrix operand order repaired and checked by reading, compiled, not run on hardware
//
#include <hip/hip_runtime.h>
#include <stdint.h>

// ---------------------------------------------------------------------------
// MI455X (gfx1250) implementation: bf16 WMMA everywhere the FLOPs are.
//   D=768, FF=384, S=2048, B=8.  Compute-bound (~170 GFLOP vs ~100MB HBM).
// ---------------------------------------------------------------------------

typedef __attribute__((ext_vector_type(16))) __bf16        v16bf;
typedef __attribute__((ext_vector_type(8)))  float         v8f;
typedef __attribute__((ext_vector_type(8)))  unsigned int  v8u;

#define D_MODEL 768
#define D_FF    384
#define SEQ     2048
#define NBATCH  8
#define NROWS   (NBATCH * SEQ)          // 16384
#define LN_EPS  1e-5f
#define QK_SCALE 0.03608439182435161f   // 1/sqrt(768)

static __device__ __forceinline__ unsigned short f2bf(float f) {
  union { float f; uint32_t u; } c; c.f = f;
  uint32_t u = c.u + 0x7FFFu + ((c.u >> 16) & 1u);  // round-to-nearest-even
  return (unsigned short)(u >> 16);
}

// A-matrix (16x32 bf16) packed-pair K offset for VGPR v, lane-half h.
// ISA 7.12.2: VGPR0..3 -> K={0..7}+8h, VGPR4..7 -> K={16..23}+8h (pairs).
static __device__ __forceinline__ int a_k0(int v, int h) {
  return (v < 4 ? 2 * v : 16 + 2 * (v - 4)) + h * 8;
}
// B-matrix (32x16 bf16): same K order as A: VGPR0..3 -> K={0..7}+8h, VGPR4..7 -> K={16..23}+8h (pairs).
static __device__ __forceinline__ int b_k0(int v, int h) {
  return (v < 4 ? 2 * v : 16 + 2 * (v - 4)) + h * 8;
}

static __device__ __forceinline__ v8f vzero8() {
  v8f z;
#pragma unroll
  for (int i = 0; i < 8; ++i) z[i] = 0.0f;
  return z;
}

static __device__ __forceinline__ v8f wmma_bf16(v8u a, v8u b, v8f c) {
  return __builtin_amdgcn_wmma_f32_16x16x32_bf16(
      false, __builtin_bit_cast(v16bf, a),
      false, __builtin_bit_cast(v16bf, b),
      (short)0, c, false, false);
}

// ---------------------------------------------------------------------------
// 0a. f32 -> bf16 convert (grid-stride)
// ---------------------------------------------------------------------------
__global__ void __launch_bounds__(256)
k_cvt(const float* __restrict__ in, unsigned short* __restrict__ out, int n) {
  for (int i = blockIdx.x * blockDim.x + threadIdx.x; i < n;
       i += gridDim.x * blockDim.x)
    out[i] = f2bf(in[i]);
}

// 0b. transpose f32 [K][N] -> bf16 [N][K]
__global__ void __launch_bounds__(256)
k_transpose(const float* __restrict__ in, unsigned short* __restrict__ out,
            int K, int N) {
  int total = K * N;
  for (int i = blockIdx.x * blockDim.x + threadIdx.x; i < total;
       i += gridDim.x * blockDim.x) {
    int nn = i / K, kk = i % K;
    out[i] = f2bf(in[kk * N + nn]);
  }
}

// ---------------------------------------------------------------------------
// 1. QKV projection: [16384,768] x [768,768] (x3 via blockIdx.z), bf16 out.
//    Each wave: 16x64 tile. block = 8 waves -> 128 rows x 64 cols.
// ---------------------------------------------------------------------------
__global__ void __launch_bounds__(256)
k_qkv(const unsigned short* __restrict__ hsbf,
      const unsigned short* __restrict__ wqT,
      const unsigned short* __restrict__ wkT,
      const unsigned short* __restrict__ wvT,
      const float* __restrict__ bq, const float* __restrict__ bk,
      const float* __restrict__ bv,
      unsigned short* __restrict__ qbf, unsigned short* __restrict__ kbf,
      unsigned short* __restrict__ vbf) {
  const int lane = threadIdx.x & 31, wave = threadIdx.x >> 5;
  const int hl = lane >> 4, ln = lane & 15;
  const int z = blockIdx.z;
  const unsigned short* wT = (z == 0) ? wqT : (z == 1) ? wkT : wvT;
  const float* bias        = (z == 0) ? bq  : (z == 1) ? bk  : bv;
  unsigned short* outp     = (z == 0) ? qbf : (z == 1) ? kbf : vbf;
  const int mrow  = blockIdx.x * 128 + wave * 16;
  const int nbase = blockIdx.y * 64;

  v8f c[4];
#pragma unroll
  for (int nt = 0; nt < 4; ++nt) c[nt] = vzero8();

  for (int kb = 0; kb < D_MODEL; kb += 32) {
    v8u au;
#pragma unroll
    for (int v = 0; v < 8; ++v)
      au[v] = *(const uint32_t*)(hsbf + (size_t)(mrow + ln) * D_MODEL + kb +
                                 a_k0(v, hl));
#pragma unroll
    for (int nt = 0; nt < 4; ++nt) {
      v8u bu;
#pragma unroll
      for (int v = 0; v < 8; ++v)
        bu[v] = *(const uint32_t*)(wT + (size_t)(nbase + nt * 16 + ln) * D_MODEL +
                                   kb + b_k0(v, hl));
      c[nt] = wmma_bf16(au, bu, c[nt]);
    }
  }
#pragma unroll
  for (int nt = 0; nt < 4; ++nt) {
    const int col = nbase + nt * 16 + ln;
    const float bb = bias[col];
#pragma unroll
    for (int r = 0; r < 8; ++r)
      outp[(size_t)(mrow + r + 8 * hl) * D_MODEL + col] = f2bf(c[nt][r] + bb);
  }
}

// ---------------------------------------------------------------------------
// 2. Flash attention + LayerNorm, per (batch, 16-query tile).
//    8 waves: scores split-K (96 dims each, ds_add_f32 reduce into LDS),
//    online softmax (1 thread/row), P@V with V^T staged in LDS.
// ---------------------------------------------------------------------------
__global__ void __launch_bounds__(256)
k_attn(const unsigned short* __restrict__ qbf,
       const unsigned short* __restrict__ kbf,
       const unsigned short* __restrict__ vbf, const int* __restrict__ am,
       const float* __restrict__ lng, const float* __restrict__ lnb,
       unsigned short* __restrict__ ctxbf) {
  __shared__ __align__(16) unsigned char smem[49152 + 2048 + 1024 + 512];
  unsigned short* vT  = (unsigned short*)smem;                 // [768][32] bf16
  float*          ctx = (float*)smem;                          // alias [16][768]
  float*          P   = (float*)(smem + 49152);                // [16][32] f32
  unsigned short* pbf = (unsigned short*)(smem + 49152 + 2048);// [16][32] bf16
  float*          st  = (float*)(smem + 49152 + 3072);
  float* m_s = st; float* l_s = st + 16; float* sf_s = st + 32;
  float* mu_s = st + 48; float* rs_s = st + 64;

  const int tid = threadIdx.x, lane = tid & 31, wave = tid >> 5;
  const int hl = lane >> 4, ln = lane & 15;
  const int b = blockIdx.y;
  const int qrow0 = b * SEQ + blockIdx.x * 16;
  const int wbase = wave * 96;   // this wave's 96 output dims

  if (tid < 16) { m_s[tid] = -1e30f; l_s[tid] = 0.0f; }

  // Hold this wave's q fragments (3 K-slices of 32) for the whole key loop.
  v8u qf[3];
#pragma unroll
  for (int ks = 0; ks < 3; ++ks)
#pragma unroll
    for (int v = 0; v < 8; ++v)
      qf[ks][v] = *(const uint32_t*)(qbf + (size_t)(qrow0 + ln) * D_MODEL +
                                     wbase + ks * 32 + a_k0(v, hl));

  v8f o[6];
#pragma unroll
  for (int nt = 0; nt < 6; ++nt) o[nt] = vzero8();

  for (int kt = 0; kt < SEQ / 32; ++kt) {
    const int kg0 = b * SEQ + kt * 32;
    __syncthreads();  // previous iter done with vT / P / pbf

    // Stage V^T tile: vT[dim][key], coalesced 16B reads.
    for (int cc = tid; cc < 32 * 96; cc += 256) {
      int key = cc / 96, d0 = (cc % 96) * 8;
      uint4 raw = *(const uint4*)(vbf + (size_t)(kg0 + key) * D_MODEL + d0);
      const unsigned short* sp = (const unsigned short*)&raw;
#pragma unroll
      for (int j = 0; j < 8; ++j) vT[(d0 + j) * 32 + key] = sp[j];
    }
    for (int i = tid; i < 512; i += 256) P[i] = 0.0f;
    __syncthreads();

    // Scores: split-K across waves, 2 key-subtiles x 3 K-steps of WMMA.
    v8f s0 = vzero8(), s1 = vzero8();
#pragma unroll
    for (int ks = 0; ks < 3; ++ks) {
      v8u kb0, kb1;
#pragma unroll
      for (int v = 0; v < 8; ++v) {
        int dim = wbase + ks * 32 + b_k0(v, hl);
        kb0[v] = *(const uint32_t*)(kbf + (size_t)(kg0 + ln) * D_MODEL + dim);
        kb1[v] = *(const uint32_t*)(kbf + (size_t)(kg0 + 16 + ln) * D_MODEL + dim);
      }
      s0 = wmma_bf16(qf[ks], kb0, s0);
      s1 = wmma_bf16(qf[ks], kb1, s1);
    }
#pragma unroll
    for (int r = 0; r < 8; ++r) {
      int row = r + 8 * hl;
      atomicAdd(&P[row * 32 + ln], s0[r]);        // ds_add_f32
      atomicAdd(&P[row * 32 + 16 + ln], s1[r]);
    }
    __syncthreads();

    // Online softmax update, one thread per query row.
    if (tid < 16) {
      float mo = m_s[tid], mx = mo;
      float xv[32];
#pragma unroll
      for (int j = 0; j < 32; ++j) {
        float x = P[tid * 32 + j] * QK_SCALE;
        if (am[b * SEQ + kt * 32 + j] == 0) x = -10000.0f;
        xv[j] = x;
        mx = fmaxf(mx, x);
      }
      float sf = __expf(mo - mx);
      float sum = 0.0f;
#pragma unroll
      for (int j = 0; j < 32; ++j) {
        float p = __expf(xv[j] - mx);
        sum += p;
        pbf[tid * 32 + j] = f2bf(p);
      }
      m_s[tid] = mx;
      l_s[tid] = l_s[tid] * sf + sum;
      sf_s[tid] = sf;
    }
    __syncthreads();

    // Rescale accumulators and P @ V (one K=32 WMMA per 16-dim tile).
    v8u pa;
#pragma unroll
    for (int v = 0; v < 8; ++v)
      pa[v] = *(const uint32_t*)(pbf + ln * 32 + a_k0(v, hl));
#pragma unroll
    for (int nt = 0; nt < 6; ++nt) {
#pragma unroll
      for (int r = 0; r < 8; ++r) o[nt][r] *= sf_s[r + 8 * hl];
      int dcol = wbase + nt * 16 + ln;
      v8u bv;
#pragma unroll
      for (int v = 0; v < 8; ++v)
        bv[v] = *(const uint32_t*)(vT + dcol * 32 + b_k0(v, hl));
      o[nt] = wmma_bf16(pa, bv, o[nt]);
    }
  }
  __syncthreads();

  // Normalize by l, dump context into LDS (aliases vT), fused LayerNorm.
#pragma unroll
  for (int nt = 0; nt < 6; ++nt)
#pragma unroll
    for (int r = 0; r < 8; ++r) {
      int row = r + 8 * hl;
      ctx[row * D_MODEL + wbase + nt * 16 + ln] = o[nt][r] / l_s[row];
    }
  __syncthreads();
  if (tid < 16) {
    float mu = 0.0f;
    for (int j = 0; j < D_MODEL; ++j) mu += ctx[tid * D_MODEL + j];
    mu /= (float)D_MODEL;
    float var = 0.0f;
    for (int j = 0; j < D_MODEL; ++j) {
      float d = ctx[tid * D_MODEL + j] - mu;
      var += d * d;
    }
    var /= (float)D_MODEL;
    mu_s[tid] = mu;
    rs_s[tid] = rsqrtf(var + LN_EPS);
  }
  __syncthreads();
  for (int i = tid; i < 16 * D_MODEL; i += 256) {
    int row = i / D_MODEL, d = i % D_MODEL;
    float x = (ctx[i] - mu_s[row]) * rs_s[row] * lng[d] + lnb[d];
    ctxbf[(size_t)(qrow0 + row) * D_MODEL + d] = f2bf(x);
  }
}

// ---------------------------------------------------------------------------
// 3. MLP: h = relu(LN2(ctx @ W1 + b1)); token score = h @ W2 + b2.
//    One block per 16 rows; 8 waves x 3 N-tiles cover all 384 cols.
// ---------------------------------------------------------------------------
__global__ void __launch_bounds__(256)
k_mlp(const unsigned short* __restrict__ ctxbf,
      const unsigned short* __restrict__ w1T, const float* __restrict__ b1,
      const float* __restrict__ g2, const float* __restrict__ bb2,
      const float* __restrict__ W2, const float* __restrict__ b2,
      float* __restrict__ tsraw) {
  __shared__ float h[16 * D_FF];
  const int tid = threadIdx.x, lane = tid & 31, wave = tid >> 5;
  const int hl = lane >> 4, ln = lane & 15;
  const int mrow = blockIdx.x * 16;
  const int nb = wave * 48;

  v8f c[3];
#pragma unroll
  for (int nt = 0; nt < 3; ++nt) c[nt] = vzero8();

  for (int kb = 0; kb < D_MODEL; kb += 32) {
    v8u au;
#pragma unroll
    for (int v = 0; v < 8; ++v)
      au[v] = *(const uint32_t*)(ctxbf + (size_t)(mrow + ln) * D_MODEL + kb +
                                 a_k0(v, hl));
#pragma unroll
    for (int nt = 0; nt < 3; ++nt) {
      v8u bu;
#pragma unroll
      for (int v = 0; v < 8; ++v)
        bu[v] = *(const uint32_t*)(w1T + (size_t)(nb + nt * 16 + ln) * D_MODEL +
                                   kb + b_k0(v, hl));
      c[nt] = wmma_bf16(au, bu, c[nt]);
    }
  }
#pragma unroll
  for (int nt = 0; nt < 3; ++nt) {
    int col = nb + nt * 16 + ln;
    float bb = b1[col];
#pragma unroll
    for (int r = 0; r < 8; ++r)
      h[(r + 8 * hl) * D_FF + col] = c[nt][r] + bb;
  }
  __syncthreads();
  if (tid < 16) {
    float mu = 0.0f;
    for (int j = 0; j < D_FF; ++j) mu += h[tid * D_FF + j];
    mu /= (float)D_FF;
    float var = 0.0f;
    for (int j = 0; j < D_FF; ++j) {
      float d = h[tid * D_FF + j] - mu;
      var += d * d;
    }
    var /= (float)D_FF;
    float rs = rsqrtf(var + LN_EPS);
    float acc = 0.0f;
    for (int j = 0; j < D_FF; ++j) {
      float x = (h[tid * D_FF + j] - mu) * rs * g2[j] + bb2[j];
      x = fmaxf(x, 0.0f);
      acc += x * W2[j];
    }
    tsraw[mrow + tid] = acc + b2[0];
  }
}

// ---------------------------------------------------------------------------
// 4. Per-batch masked softmax, threshold, attended states, pooled mean.
// ---------------------------------------------------------------------------
__global__ void __launch_bounds__(256)
k_final(const float* __restrict__ tsraw, const int* __restrict__ am,
        const float* __restrict__ hs, float* __restrict__ out) {
  __shared__ float sc[SEQ];
  __shared__ float rm[SEQ];
  __shared__ float red[256];
  float* o_ts = out;
  float* o_tp = out + NROWS;
  float* o_rm = out + 2 * NROWS;
  float* o_att = out + 3 * NROWS;
  float* o_pool = o_att + (size_t)NROWS * D_MODEL;

  const int b = blockIdx.x, tid = threadIdx.x;
  float lmax = -1e30f, lvalid = 0.0f;
  for (int s = tid; s < SEQ; s += 256) {
    float x = tsraw[b * SEQ + s];
    int mv = am[b * SEQ + s];
    lvalid += (float)mv;
    if (mv == 0) x = -10000.0f;
    sc[s] = x;
    o_ts[b * SEQ + s] = x;
    lmax = fmaxf(lmax, x);
  }
  red[tid] = lmax; __syncthreads();
  for (int s2 = 128; s2 > 0; s2 >>= 1) {
    if (tid < s2) red[tid] = fmaxf(red[tid], red[tid + s2]);
    __syncthreads();
  }
  float mx = red[0]; __syncthreads();

  float lsum = 0.0f;
  for (int s = tid; s < SEQ; s += 256) {
    float e = __expf(sc[s] - mx);
    sc[s] = e;
    lsum += e;
  }
  red[tid] = lsum; __syncthreads();
  for (int s2 = 128; s2 > 0; s2 >>= 1) {
    if (tid < s2) red[tid] += red[tid + s2];
    __syncthreads();
  }
  float inv = 1.0f / red[0]; __syncthreads();

  red[tid] = lvalid; __syncthreads();
  for (int s2 = 128; s2 > 0; s2 >>= 1) {
    if (tid < s2) red[tid] += red[tid + s2];
    __syncthreads();
  }
  float valid = red[0]; __syncthreads();

  for (int s = tid; s < SEQ; s += 256) {
    float p = sc[s] * inv;
    o_tp[b * SEQ + s] = p;
    float m = (p > 0.5f) ? 1.0f : 0.0f;
    rm[s] = m;
    o_rm[b * SEQ + s] = m;
  }
  __syncthreads();

  const size_t base = (size_t)b * SEQ * D_MODEL;
  for (size_t i = tid; i < (size_t)SEQ * D_MODEL; i += 256) {
    int s = (int)(i / D_MODEL);
    o_att[base + i] = hs[base + i] * rm[s];
  }
  for (int d = tid; d < D_MODEL; d += 256) {
    float acc = 0.0f;
    for (int s = 0; s < SEQ; ++s)
      acc += hs[base + (size_t)s * D_MODEL + d] * rm[s];
    o_pool[b * D_MODEL + d] = acc / valid;
  }
}

// ---------------------------------------------------------------------------
extern "C" void kernel_launch(void* const* d_in, const int* in_sizes, int n_in,
                              void* d_out, int out_size, void* d_ws,
                              size_t ws_size, hipStream_t stream) {
  (void)in_sizes; (void)n_in; (void)out_size; (void)ws_size;
  const float* hs  = (const float*)d_in[0];
  const int*   am  = (const int*)d_in[1];
  const float* Wq  = (const float*)d_in[2];  const float* bq  = (const float*)d_in[3];
  const float* Wk  = (const float*)d_in[4];  const float* bk  = (const float*)d_in[5];
  const float* Wv  = (const float*)d_in[6];  const float* bv  = (const float*)d_in[7];
  const float* lng = (const float*)d_in[8];  const float* lnb = (const float*)d_in[9];
  const float* W1  = (const float*)d_in[10]; const float* b1  = (const float*)d_in[11];
  const float* g2  = (const float*)d_in[12]; const float* bb2 = (const float*)d_in[13];
  const float* W2  = (const float*)d_in[14]; const float* b2  = (const float*)d_in[15];

  char* w = (char*)d_ws;
  size_t off = 0;
  auto alloc = [&](size_t bytes) -> void* {
    void* p = w + off;
    off += (bytes + 255) & ~(size_t)255;
    return p;
  };
  unsigned short* hsbf  = (unsigned short*)alloc((size_t)NROWS * D_MODEL * 2);
  unsigned short* qbf   = (unsigned short*)alloc((size_t)NROWS * D_MODEL * 2);
  unsigned short* kbf   = (unsigned short*)alloc((size_t)NROWS * D_MODEL * 2);
  unsigned short* vbf   = (unsigned short*)alloc((size_t)NROWS * D_MODEL * 2);
  unsigned short* ctxbf = (unsigned short*)alloc((size_t)NROWS * D_MODEL * 2);
  unsigned short* wqT   = (unsigned short*)alloc((size_t)D_MODEL * D_MODEL * 2);
  unsigned short* wkT   = (unsigned short*)alloc((size_t)D_MODEL * D_MODEL * 2);
  unsigned short* wvT   = (unsigned short*)alloc((size_t)D_MODEL * D_MODEL * 2);
  unsigned short* w1T   = (unsigned short*)alloc((size_t)D_FF * D_MODEL * 2);
  float*          tsraw = (float*)alloc((size_t)NROWS * 4);

  k_cvt<<<512, 256, 0, stream>>>(hs, hsbf, NROWS * D_MODEL);
  k_transpose<<<256, 256, 0, stream>>>(Wq, wqT, D_MODEL, D_MODEL);
  k_transpose<<<256, 256, 0, stream>>>(Wk, wkT, D_MODEL, D_MODEL);
  k_transpose<<<256, 256, 0, stream>>>(Wv, wvT, D_MODEL, D_MODEL);
  k_transpose<<<256, 256, 0, stream>>>(W1, w1T, D_MODEL, D_FF);

  k_qkv<<<dim3(NROWS / 128, D_MODEL / 64, 3), 256, 0, stream>>>(
      hsbf, wqT, wkT, wvT, bq, bk, bv, qbf, kbf, vbf);
  k_attn<<<dim3(SEQ / 16, NBATCH), 256, 0, stream>>>(
      qbf, kbf, vbf, am, lng, lnb, ctxbf);
  k_mlp<<<NROWS / 16, 256, 0, stream>>>(ctxbf, w1T, b1, g2, bb2, W2, b2, tsraw);
  k_final<<<NBATCH, 256, 0, stream>>>(tsraw, am, hs, (float*)d_out);
}
